// SparseNeuralNetwork_44504451121689
// MI455X (gfx1250) — matrix operand order repaired and checked by reading, not compiled, not run on hardware
//
#include <hip/hip_runtime.h>
#include <hip/hip_bf16.h>

// Problem constants (match reference)
#define BN    8192      // batch
#define DN    1024      // features
#define H1N   8
#define H2N   4
#define C1N   (DN*H1N)  // 8192 layer-1 channels
#define C2N   (DN*H2N)  // 4096 layer-2 channels
#define OUTN  256
#define EPSF  1e-5f

typedef __attribute__((ext_vector_type(16))) _Float16 v16h;
typedef __attribute__((ext_vector_type(8)))  _Float16 v8h;
typedef __attribute__((ext_vector_type(4)))  _Float16 v4h;
typedef __attribute__((ext_vector_type(8)))  float    v8f;

__device__ __forceinline__ v16h ld_frag16(const _Float16* lo, const _Float16* hi) {
  v8h a = *(const v8h*)lo;
  v8h b = *(const v8h*)hi;
  return __builtin_shufflevector(a, b, 0,1,2,3,4,5,6,7,8,9,10,11,12,13,14,15);
}

// ---------------------------------------------------------------------------
// 1) Column stats of x: sum and sumsq over batch, per feature d (1024 each)
//    grid 64 blocks x 256 thr; block covers 128 rows, thread owns 4 columns
// ---------------------------------------------------------------------------
__global__ __launch_bounds__(256) void k_xstats(const float* __restrict__ x,
                                                float* __restrict__ xsum,
                                                float* __restrict__ xsq) {
  const int t  = threadIdx.x;
  const int r0 = blockIdx.x * 128;
  float s[4] = {0.f,0.f,0.f,0.f}, q[4] = {0.f,0.f,0.f,0.f};
  for (int rr = 0; rr < 128; ++rr) {
    const float* row = x + (size_t)(r0 + rr) * DN;
#pragma unroll
    for (int j = 0; j < 4; ++j) {
      float v = row[t + j*256];
      s[j] += v;
      q[j] = fmaf(v, v, q[j]);
    }
  }
#pragma unroll
  for (int j = 0; j < 4; ++j) {
    atomicAdd(&xsum[t + j*256], s[j]);
    atomicAdd(&xsq [t + j*256], q[j]);
  }
}

// ---------------------------------------------------------------------------
// 2) Fold layer-1 (scale W1 + BN + stats) into per-channel affine A1,C1
//    bn1(relu input) = A1[c]*x + C1[c];  b1 cancels under train-mode BN
// ---------------------------------------------------------------------------
__global__ __launch_bounds__(256) void k_fold1(const float* __restrict__ W1,
                                               const float* __restrict__ g1,
                                               const float* __restrict__ be1,
                                               const float* __restrict__ xsum,
                                               const float* __restrict__ xsq,
                                               float* __restrict__ A1,
                                               float* __restrict__ C1) {
  const int c = blockIdx.x * 256 + threadIdx.x;   // 0..8191
  const int d = c >> 3;
  const float invB = 1.0f / (float)BN;
  float mu  = xsum[d] * invB;
  float var = fmaf(-mu, mu, xsq[d] * invB);
  float w   = W1[c];
  float a   = g1[c] * w * rsqrtf(fmaf(w*w, var, EPSF));
  A1[c] = a;
  C1[c] = be1[c] - a * mu;
}

// ---------------------------------------------------------------------------
// 3) Batch stats of layer-2 pre-BN (b2 excluded; it cancels under BN).
//    grid 128 blocks x 256 thr; block covers 64 rows; thread owns 4 d's
//    (one at a time) -> 8 register accumulators, then atomics.
// ---------------------------------------------------------------------------
__global__ __launch_bounds__(256) void k_h2stats(const float* __restrict__ x,
                                                 const float* __restrict__ W2,
                                                 const float* __restrict__ A1,
                                                 const float* __restrict__ C1,
                                                 float* __restrict__ s2sum,
                                                 float* __restrict__ s2sq) {
  const int t  = threadIdx.x;
  const int r0 = blockIdx.x * 64;
  for (int dit = 0; dit < 4; ++dit) {
    const int d = dit * 256 + t;
    float w2[H2N][H1N], a1[H1N], c1[H1N];
#pragma unroll
    for (int o = 0; o < H2N; ++o)
#pragma unroll
      for (int i = 0; i < H1N; ++i) w2[o][i] = W2[d*32 + o*8 + i];
#pragma unroll
    for (int i = 0; i < H1N; ++i) { a1[i] = A1[d*8 + i]; c1[i] = C1[d*8 + i]; }
    float s[H2N] = {0.f,0.f,0.f,0.f}, q[H2N] = {0.f,0.f,0.f,0.f};
    for (int rr = 0; rr < 64; ++rr) {
      float xv = x[(size_t)(r0 + rr) * DN + d];
      float tv[H1N];
#pragma unroll
      for (int i = 0; i < H1N; ++i) {
        float v = fmaf(a1[i], xv, c1[i]);
        tv[i] = v > 0.f ? v : 0.f;
      }
#pragma unroll
      for (int o = 0; o < H2N; ++o) {
        float h = 0.f;
#pragma unroll
        for (int i = 0; i < H1N; ++i) h = fmaf(w2[o][i], tv[i], h);
        s[o] += h;
        q[o] = fmaf(h, h, q[o]);
      }
    }
#pragma unroll
    for (int o = 0; o < H2N; ++o) {
      atomicAdd(&s2sum[d*4 + o], s[o]);
      atomicAdd(&s2sq [d*4 + o], q[o]);
    }
  }
}

// ---------------------------------------------------------------------------
// 4) Fold layer-2 BN into affine A2,C2 per channel (4096)
// ---------------------------------------------------------------------------
__global__ __launch_bounds__(256) void k_fold2(const float* __restrict__ g2,
                                               const float* __restrict__ be2,
                                               const float* __restrict__ s2sum,
                                               const float* __restrict__ s2sq,
                                               float* __restrict__ A2,
                                               float* __restrict__ C2) {
  const int c = blockIdx.x * 256 + threadIdx.x;   // 0..4095
  const float invB = 1.0f / (float)BN;
  float mu  = s2sum[c] * invB;
  float var = fmaf(-mu, mu, s2sq[c] * invB);
  float a   = g2[c] * rsqrtf(var + EPSF);
  A2[c] = a;
  C2[c] = be2[c] - a * mu;
}

// ---------------------------------------------------------------------------
// 5) Materialize H2 = relu(A2*h2pre + C2) as fp16, row-major (8192 x 4096)
//    grid (4, 256) x 256 thr: block = 256 features x 32 rows
// ---------------------------------------------------------------------------
__global__ __launch_bounds__(256) void k_h2mat(const float* __restrict__ x,
                                               const float* __restrict__ W2,
                                               const float* __restrict__ A1,
                                               const float* __restrict__ C1,
                                               const float* __restrict__ A2,
                                               const float* __restrict__ C2,
                                               _Float16* __restrict__ H2) {
  const int t  = threadIdx.x;
  const int d  = blockIdx.x * 256 + t;     // 0..1023
  const int r0 = blockIdx.y * 32;
  float w2[H2N][H1N], a1[H1N], c1[H1N], a2[H2N], c2[H2N];
#pragma unroll
  for (int o = 0; o < H2N; ++o)
#pragma unroll
    for (int i = 0; i < H1N; ++i) w2[o][i] = W2[d*32 + o*8 + i];
#pragma unroll
  for (int i = 0; i < H1N; ++i) { a1[i] = A1[d*8 + i]; c1[i] = C1[d*8 + i]; }
#pragma unroll
  for (int o = 0; o < H2N; ++o) { a2[o] = A2[d*4 + o]; c2[o] = C2[d*4 + o]; }
  for (int rr = 0; rr < 32; ++rr) {
    float xv = x[(size_t)(r0 + rr) * DN + d];
    float tv[H1N];
#pragma unroll
    for (int i = 0; i < H1N; ++i) {
      float v = fmaf(a1[i], xv, c1[i]);
      tv[i] = v > 0.f ? v : 0.f;
    }
    v4h o4;
#pragma unroll
    for (int o = 0; o < H2N; ++o) {
      float h = 0.f;
#pragma unroll
      for (int i = 0; i < H1N; ++i) h = fmaf(w2[o][i], tv[i], h);
      float y = fmaf(a2[o], h, c2[o]);
      o4[o] = (_Float16)(y > 0.f ? y : 0.f);
    }
    *(v4h*)(H2 + (size_t)(r0 + rr) * C2N + d*4) = o4;
  }
}

// ---------------------------------------------------------------------------
// 6) Convert Wfc (256 x 4096 f32) -> WfcT (256 x 4096 f16), same layout, no transpose
// ---------------------------------------------------------------------------
__global__ __launch_bounds__(256) void k_wfct(const float* __restrict__ Wfc,
                                              _Float16* __restrict__ WfcT) {
  const int id = blockIdx.x * 256 + threadIdx.x;  // 0 .. 4096*256-1
  const int k = id >> 8;
  const int n = id & 255;
  WfcT[(size_t)n * C2N + k] = (_Float16)Wfc[(size_t)n * C2N + k];
}

// ---------------------------------------------------------------------------
// 7) FC head: out(8192x256) = H2(8192x4096,f16) x WfcT^T (WfcT 256x4096,f16) + bfc
//    wave32 WMMA f32_16x16x32_f16; each wave computes a 32x32 tile.
//    A frag: lane<16 row=lane K{0..7,16..23}; lane>=16 row=lane-16 K{8..15,24..31}
//    B frag: lane<16 col=lane K{0..7,16..23}; lane>=16 col=lane-16 K{8..15,24..31}
//    C/D:    VGPR j -> row j+8*(lane>=16), col lane&15
// ---------------------------------------------------------------------------
__global__ __launch_bounds__(256) void k_fc(const _Float16* __restrict__ H2,
                                            const _Float16* __restrict__ WfcT,
                                            const float* __restrict__ bfc,
                                            float* __restrict__ out) {
  const int wave = threadIdx.x >> 5;
  const int lane = threadIdx.x & 31;
  const int gw   = blockIdx.x * 8 + wave;        // 0..2047
  const int n0   = (gw & 7) * 32;                // 8 N-tiles of 32
  const int m0   = (gw >> 3) * 32;               // 256 M-tiles of 32
  const int half = lane >> 4;
  const int r    = lane & 15;

  const _Float16* pa = H2 + (size_t)(m0 + r) * C2N + half * 8;
  const _Float16* pb = WfcT + (size_t)(n0 + r) * C2N + half * 8;

  v8f acc00 = {}, acc01 = {}, acc10 = {}, acc11 = {};
  for (int k = 0; k < C2N; k += 32) {
    v16h a0 = ld_frag16(pa + k,             pa + k + 16);
    v16h a1 = ld_frag16(pa + k + 16*C2N,    pa + k + 16*C2N + 16);
    const _Float16* pbk = pb + k;
    v16h b0 = ld_frag16(pbk,      pbk + 16);
    v16h b1 = ld_frag16(pbk + 16*C2N, pbk + 16*C2N + 16);
    acc00 = __builtin_amdgcn_wmma_f32_16x16x32_f16(false, a0, false, b0, (short)0, acc00, false, false);
    acc01 = __builtin_amdgcn_wmma_f32_16x16x32_f16(false, a0, false, b1, (short)0, acc01, false, false);
    acc10 = __builtin_amdgcn_wmma_f32_16x16x32_f16(false, a1, false, b0, (short)0, acc10, false, false);
    acc11 = __builtin_amdgcn_wmma_f32_16x16x32_f16(false, a1, false, b1, (short)0, acc11, false, false);
  }

  const float bias0 = bfc[n0 + r];
  const float bias1 = bfc[n0 + 16 + r];
  float* o0 = out + (size_t)(m0 + half * 8) * OUTN + n0 + r;
#pragma unroll
  for (int j = 0; j < 8; ++j) {
    o0[(size_t)j * OUTN]             = acc00[j] + bias0;
    o0[(size_t)j * OUTN + 16]        = acc01[j] + bias1;
    o0[(size_t)(j + 16) * OUTN]      = acc10[j] + bias0;
    o0[(size_t)(j + 16) * OUTN + 16] = acc11[j] + bias1;
  }
}

// ---------------------------------------------------------------------------
// Launch
// ---------------------------------------------------------------------------
extern "C" void kernel_launch(void* const* d_in, const int* in_sizes, int n_in,
                              void* d_out, int out_size, void* d_ws, size_t ws_size,
                              hipStream_t stream) {
  const float* x   = (const float*)d_in[0];
  const float* W1  = (const float*)d_in[1];
  // const float* b1 = (const float*)d_in[2];  // cancels under train-mode BN
  const float* g1  = (const float*)d_in[3];
  const float* be1 = (const float*)d_in[4];
  const float* W2  = (const float*)d_in[5];
  // const float* b2 = (const float*)d_in[6];  // cancels under train-mode BN
  const float* g2  = (const float*)d_in[7];
  const float* be2 = (const float*)d_in[8];
  const float* Wfc = (const float*)d_in[9];
  const float* bfc = (const float*)d_in[10];
  float* out = (float*)d_out;

  // Workspace layout
  float* ws   = (float*)d_ws;
  float* xsum  = ws;            // 1024
  float* xsq   = ws + 1024;     // 1024
  float* s2sum = ws + 2048;     // 4096
  float* s2sq  = ws + 6144;     // 4096
  float* A1    = ws + 10240;    // 8192
  float* C1    = ws + 18432;    // 8192
  float* A2    = ws + 26624;    // 4096
  float* C2    = ws + 30720;    // 4096
  char* base = (char*)d_ws;
  _Float16* H2   = (_Float16*)(base + (size_t)(1 << 18));                        // 64 MB
  _Float16* WfcT = (_Float16*)(base + (size_t)(1 << 18) + (size_t)BN * C2N * 2); // 2 MB

  // Zero the atomic-accumulated stats regions (first 10240 floats)
  hipMemsetAsync(d_ws, 0, 10240 * sizeof(float), stream);

  k_xstats <<<64,  256, 0, stream>>>(x, xsum, xsq);
  k_fold1  <<<32,  256, 0, stream>>>(W1, g1, be1, xsum, xsq, A1, C1);
  k_h2stats<<<128, 256, 0, stream>>>(x, W2, A1, C1, s2sum, s2sq);
  k_fold2  <<<16,  256, 0, stream>>>(g2, be2, s2sum, s2sq, A2, C2);
  k_h2mat  <<<dim3(4, 256), 256, 0, stream>>>(x, W2, A1, C1, A2, C2, H2);
  k_wfct   <<<4096, 256, 0, stream>>>(Wfc, WfcT);
  k_fc     <<<256, 256, 0, stream>>>(H2, WfcT, bfc, out);
}
